// MultiHeadAttention_57836029608452
// MI455X (gfx1250) — matrix operand order repaired and checked by reading, not compiled, not run on hardware
//
#include <hip/hip_runtime.h>
#include <hip/hip_bf16.h>

typedef __attribute__((ext_vector_type(16))) __bf16 v16bf;
typedef __attribute__((ext_vector_type(8)))  __bf16 v8bf;
typedef __attribute__((ext_vector_type(8)))  float  v8f;
typedef unsigned short u16;

#define DMODEL 1024
#define NHEADS 16
#define HDIM   64
#define SEQ    2048
#define BATCH  2
#define MROWS  (BATCH * SEQ)          // 4096

// ---------- helpers ----------
__device__ __forceinline__ u16 f2bf(float f) {
    union { float f; unsigned u; } v; v.f = f;
    unsigned u = v.u;
    return (u16)((u + 0x7FFFu + ((u >> 16) & 1u)) >> 16);   // RNE
}

// A-fragment 16x32 bf16: lane m = l&15; K = k0 + (l>>4)*16 + e (16 contiguous),
// matching load_bfrag
__device__ __forceinline__ v16bf load_afrag(const u16* __restrict__ src, int m, int k0,
                                            int ld, int lane) {
    int base = k0 + ((lane >> 4) << 4);
    const v8bf* p0 = (const v8bf*)(src + m * ld + base);
    const v8bf* p1 = (const v8bf*)(src + m * ld + base + 8);
    v8bf lo = *p0, hi = *p1;
    v16bf a;
#pragma unroll
    for (int i = 0; i < 8; ++i) { a[i] = lo[i]; a[i + 8] = hi[i]; }
    return a;
}

// B-fragment 32x16 bf16: lane n = l&15; K = k0 + (l>>4)*16 + e (16 contiguous).
// src is row-major with column n of B stored as row n of src (B[k][n] = src[n][k]).
__device__ __forceinline__ v16bf load_bfrag(const u16* __restrict__ src, int n, int k0,
                                            int ld, int lane) {
    int kk = k0 + ((lane >> 4) << 4);
    return *(const v16bf*)(src + n * ld + kk);
}

// ---------- fp32 -> bf16 convert ----------
__global__ __launch_bounds__(256) void cvt_bf16(const float* __restrict__ src,
                                                u16* __restrict__ dst, int n) {
    int i = blockIdx.x * blockDim.x + threadIdx.x;
    if (i < n) dst[i] = f2bf(src[i]);
}

// ---------- GEMM: C[M,N] = A[M,K] @ W[N,K]^T + bias ----------
// Wave tile: 32(M) x 64(N). All 6 fragment loads issued up front, then an
// 8-WMMA chain, so the scheduler can overlap tail loads with early WMMAs.
// mode 0: bf16 out scattered to [B,H,S,Dh]
// mode 1: fp32 out row-major [M,N]  (final projection -> d_out)
// mode 2: bf16 out scattered to [B,H,Dh,S] (V transposed for PV WMMA)
__global__ __launch_bounds__(128)
void gemm_bt(const u16* __restrict__ A, const u16* __restrict__ W,
             const float* __restrict__ bias, void* __restrict__ out,
             int M, int N, int K, int mode) {
    const int lane = threadIdx.x & 31;
    const int wave = threadIdx.x >> 5;
    const int mbase = blockIdx.y * 128 + wave * 32;
    const int nbase = blockIdx.x * 64;
    const int mrow0 = mbase + (lane & 15);
    const int mrow1 = mrow0 + 16;

    v8f acc[2][4];
#pragma unroll
    for (int ms = 0; ms < 2; ++ms)
#pragma unroll
        for (int ns = 0; ns < 4; ++ns)
            acc[ms][ns] = (v8f){0.f,0.f,0.f,0.f,0.f,0.f,0.f,0.f};

    for (int k0 = 0; k0 < K; k0 += 32) {
        // issue every load of this K-step first
        v16bf a0 = load_afrag(A, mrow0, k0, K, lane);
        v16bf a1 = load_afrag(A, mrow1, k0, K, lane);
        v16bf b[4];
#pragma unroll
        for (int ns = 0; ns < 4; ++ns)
            b[ns] = load_bfrag(W, nbase + ns * 16 + (lane & 15), k0, K, lane);
        if (k0 + 32 < K) {   // prefetch next A K-chunk (-> global_prefetch_b8)
            __builtin_prefetch(A + mrow0 * K + k0 + 32, 0, 3);
            __builtin_prefetch(A + mrow1 * K + k0 + 32, 0, 3);
        }
        // then the 8-WMMA chain
#pragma unroll
        for (int ns = 0; ns < 4; ++ns) {
            acc[0][ns] = __builtin_amdgcn_wmma_f32_16x16x32_bf16(
                             false, a0, false, b[ns], (short)0, acc[0][ns], false, false);
            acc[1][ns] = __builtin_amdgcn_wmma_f32_16x16x32_bf16(
                             false, a1, false, b[ns], (short)0, acc[1][ns], false, false);
        }
    }

    const int hl = lane >> 4;
#pragma unroll
    for (int ms = 0; ms < 2; ++ms)
#pragma unroll
    for (int ns = 0; ns < 4; ++ns) {
        int n = nbase + ns * 16 + (lane & 15);
        float bv = bias[n];
#pragma unroll
        for (int v = 0; v < 8; ++v) {
            int mm = mbase + ms * 16 + v + 8 * hl;    // C layout: M = v + 8*(l>>4)
            float val = acc[ms][ns][v] + bv;
            if (mode == 1) {
                ((float*)out)[mm * N + n] = val;
            } else {
                int b2 = mm / SEQ, s = mm % SEQ;
                int h = n / HDIM, dh = n % HDIM;
                int idx = (mode == 0)
                    ? ((b2 * NHEADS + h) * SEQ + s) * HDIM + dh
                    : ((b2 * NHEADS + h) * HDIM + dh) * SEQ + s;
                ((u16*)out)[idx] = f2bf(val);
            }
        }
    }
}

// ---------- Flash attention: softmax(Q K^T / 8) V ----------
// Q,K: [B*H, S, 64] bf16;  Vt: [B*H, 64, S] bf16;  Octx: [B*S, 1024] bf16
__global__ __launch_bounds__(32)
void attn_flash(const u16* __restrict__ Q, const u16* __restrict__ Kmat,
                const u16* __restrict__ Vt, u16* __restrict__ Octx) {
    __shared__ __align__(32) u16 Pl[16 * 32];        // P tile: C-layout -> A-layout
    const int lane = threadIdx.x;
    const int qt = (blockIdx.x % (SEQ / 16)) * 16;
    const int bh = blockIdx.x / (SEQ / 16);
    const u16* Qb = Q    + (size_t)bh * SEQ * HDIM;
    const u16* Kb = Kmat + (size_t)bh * SEQ * HDIM;
    const u16* Vb = Vt   + (size_t)bh * HDIM * SEQ;

    const int mrow = qt + (lane & 15);
    v16bf qa0 = load_afrag(Qb, mrow, 0, HDIM, lane);
    v16bf qa1 = load_afrag(Qb, mrow, 32, HDIM, lane);

    v8f o[4];
#pragma unroll
    for (int i = 0; i < 4; ++i) o[i] = (v8f){0.f,0.f,0.f,0.f,0.f,0.f,0.f,0.f};
    float rmax[8], rsum[8];
#pragma unroll
    for (int v = 0; v < 8; ++v) { rmax[v] = -1e30f; rsum[v] = 0.f; }

    const float c = 0.125f * 1.44269504088896f;      // (1/sqrt(64)) * log2(e)
    const v8f z = (v8f){0.f,0.f,0.f,0.f,0.f,0.f,0.f,0.f};

    for (int kt = 0; kt < SEQ; kt += 32) {
        // all 4 K-fragment loads first, then the 4 score WMMAs
        v16bf kb[4];
        kb[0] = load_bfrag(Kb, kt +      (lane & 15), 0,  HDIM, lane);
        kb[1] = load_bfrag(Kb, kt +      (lane & 15), 32, HDIM, lane);
        kb[2] = load_bfrag(Kb, kt + 16 + (lane & 15), 0,  HDIM, lane);
        kb[3] = load_bfrag(Kb, kt + 16 + (lane & 15), 32, HDIM, lane);
        v8f s0 = z, s1 = z;
        s0 = __builtin_amdgcn_wmma_f32_16x16x32_bf16(false, qa0, false, kb[0], (short)0, s0, false, false);
        s0 = __builtin_amdgcn_wmma_f32_16x16x32_bf16(false, qa1, false, kb[1], (short)0, s0, false, false);
        s1 = __builtin_amdgcn_wmma_f32_16x16x32_bf16(false, qa0, false, kb[2], (short)0, s1, false, false);
        s1 = __builtin_amdgcn_wmma_f32_16x16x32_bf16(false, qa1, false, kb[3], (short)0, s1, false, false);

        // online softmax per row (row spans 16 lanes of one accumulator slot)
#pragma unroll
        for (int v = 0; v < 8; ++v) {
            float mx = fmaxf(s0[v], s1[v]);
#pragma unroll
            for (int d = 1; d < 16; d <<= 1) mx = fmaxf(mx, __shfl_xor(mx, d, 16));
            float mnew = fmaxf(rmax[v], mx);
            float corr = exp2f((rmax[v] - mnew) * c);
            rmax[v] = mnew;
            float p0 = exp2f((s0[v] - mnew) * c);
            float p1 = exp2f((s1[v] - mnew) * c);
            float ps = p0 + p1;
#pragma unroll
            for (int d = 1; d < 16; d <<= 1) ps += __shfl_xor(ps, d, 16);
            rsum[v] = rsum[v] * corr + ps;
            o[0][v] *= corr; o[1][v] *= corr; o[2][v] *= corr; o[3][v] *= corr;
            int mloc = v + ((lane >> 4) << 3);
            Pl[mloc * 32 +      (lane & 15)] = f2bf(p0);
            Pl[mloc * 32 + 16 + (lane & 15)] = f2bf(p1);
        }
        __syncthreads();
        // V fragments + P A-fragment issued before the PV WMMA chain
        v16bf vb[4];
#pragma unroll
        for (int ns = 0; ns < 4; ++ns)
            vb[ns] = load_bfrag(Vb, ns * 16 + (lane & 15), kt, SEQ, lane);
        v16bf pa = load_afrag(Pl, lane & 15, 0, 32, lane);   // P as 16x32 A-fragment
#pragma unroll
        for (int ns = 0; ns < 4; ++ns)
            o[ns] = __builtin_amdgcn_wmma_f32_16x16x32_bf16(
                        false, pa, false, vb[ns], (short)0, o[ns], false, false);
        __syncthreads();
    }

    const int b = bh / NHEADS, h = bh % NHEADS;
#pragma unroll
    for (int ns = 0; ns < 4; ++ns)
#pragma unroll
        for (int v = 0; v < 8; ++v) {
            int mloc = v + ((lane >> 4) << 3);
            int q = qt + mloc;
            int n = h * HDIM + ns * 16 + (lane & 15);
            Octx[(size_t)(b * SEQ + q) * DMODEL + n] = f2bf(o[ns][v] / rsum[v]);
        }
}

// ---------- host ----------
extern "C" void kernel_launch(void* const* d_in, const int* in_sizes, int n_in,
                              void* d_out, int out_size, void* d_ws, size_t ws_size,
                              hipStream_t stream) {
    (void)in_sizes; (void)n_in; (void)out_size; (void)ws_size;
    const float* xq = (const float*)d_in[0];
    const float* xk = (const float*)d_in[1];
    const float* xv = (const float*)d_in[2];
    const float* Wq = (const float*)d_in[3];  const float* bq = (const float*)d_in[4];
    const float* Wk = (const float*)d_in[5];  const float* bk = (const float*)d_in[6];
    const float* Wv = (const float*)d_in[7];  const float* bv = (const float*)d_in[8];
    const float* Wo = (const float*)d_in[9];  const float* bo = (const float*)d_in[10];

    const int NX = MROWS * DMODEL;        // 4 Mi elems
    const int NW = DMODEL * DMODEL;       // 1 Mi elems
    char* w = (char*)d_ws;                // 56 MB total (Octx aliases xq_bf)
    const size_t MB = 1u << 20;
    u16* xq_bf = (u16*)(w + 0 * MB);      // 8 MB, reused as Octx after projections
    u16* xk_bf = (u16*)(w + 8 * MB);
    u16* xv_bf = (u16*)(w + 16 * MB);
    u16* wq_bf = (u16*)(w + 24 * MB);
    u16* wk_bf = (u16*)(w + 26 * MB);
    u16* wv_bf = (u16*)(w + 28 * MB);
    u16* wo_bf = (u16*)(w + 30 * MB);
    u16* Qb    = (u16*)(w + 32 * MB);     // [B*H, S, 64]
    u16* Kb    = (u16*)(w + 40 * MB);     // [B*H, S, 64]
    u16* Vtb   = (u16*)(w + 48 * MB);     // [B*H, 64, S]
    u16* Octx  = xq_bf;                   // [B*S, 1024]

    hipLaunchKernelGGL(cvt_bf16, dim3((NX + 255) / 256), dim3(256), 0, stream, xq, xq_bf, NX);
    hipLaunchKernelGGL(cvt_bf16, dim3((NX + 255) / 256), dim3(256), 0, stream, xk, xk_bf, NX);
    hipLaunchKernelGGL(cvt_bf16, dim3((NX + 255) / 256), dim3(256), 0, stream, xv, xv_bf, NX);
    hipLaunchKernelGGL(cvt_bf16, dim3((NW + 255) / 256), dim3(256), 0, stream, Wq, wq_bf, NW);
    hipLaunchKernelGGL(cvt_bf16, dim3((NW + 255) / 256), dim3(256), 0, stream, Wk, wk_bf, NW);
    hipLaunchKernelGGL(cvt_bf16, dim3((NW + 255) / 256), dim3(256), 0, stream, Wv, wv_bf, NW);
    hipLaunchKernelGGL(cvt_bf16, dim3((NW + 255) / 256), dim3(256), 0, stream, Wo, wo_bf, NW);

    dim3 gblk(128), ggrid(DMODEL / 64, MROWS / 128);   // (16, 32)
    hipLaunchKernelGGL(gemm_bt, ggrid, gblk, 0, stream, xq_bf, wq_bf, bq, (void*)Qb,
                       MROWS, DMODEL, DMODEL, 0);
    hipLaunchKernelGGL(gemm_bt, ggrid, gblk, 0, stream, xk_bf, wk_bf, bk, (void*)Kb,
                       MROWS, DMODEL, DMODEL, 0);
    hipLaunchKernelGGL(gemm_bt, ggrid, gblk, 0, stream, xv_bf, wv_bf, bv, (void*)Vtb,
                       MROWS, DMODEL, DMODEL, 2);

    hipLaunchKernelGGL(attn_flash, dim3(BATCH * NHEADS * (SEQ / 16)), dim3(32), 0, stream,
                       Qb, Kb, Vtb, Octx);

    hipLaunchKernelGGL(gemm_bt, ggrid, gblk, 0, stream, Octx, wo_bf, bo, d_out,
                       MROWS, DMODEL, DMODEL, 1);
}
